// BiRWKV7TimeMix_34677565948645
// MI455X (gfx1250) — matrix operand order repaired and checked by reading, not compiled, not run on hardware
//
#include <hip/hip_runtime.h>
#include <hip/hip_bf16.h>
#include <math.h>

// ---------------------------------------------------------------------------
// BiRWKV7 TimeMix forward for MI455X (gfx1250, wave32, WMMA bf16 path)
// B=4, T=1024, C=512, H=8, N=64
// All GEMM operands are pre-converted to bf16 (weights once per call, the
// small MLP weights transposed to (N,K)), so the WMMA inner loop is pure
// bf16 loads + v_wmma with no conversion chains.
// ---------------------------------------------------------------------------

#define BB 4
#define TT 1024
#define CC 512
#define HH 8
#define NN 64
#define BT (BB * TT)                 // 4096 tokens
#define EPS_GN 0.00064f

typedef __attribute__((ext_vector_type(16))) __bf16 v16bf;
typedef __attribute__((ext_vector_type(8)))  __bf16 bf16x8;
typedef __attribute__((ext_vector_type(8)))  float  v8f;

__device__ __forceinline__ __bf16 f2bf(float f) { return (__bf16)f; }

__device__ __forceinline__ float sigmoidf_(float x) {
    return 1.0f / (1.0f + expf(-x));
}

__device__ __forceinline__ float wave_sum(float v) {
#pragma unroll
    for (int off = 16; off > 0; off >>= 1) v += __shfl_xor(v, off, 32);
    return v;
}

// ---------------------------------------------------------------------------
// WMMA GEMM: Y[M,N] = X[M,K] @ W^T, X bf16 (M,K) row-major, W bf16 (N,K)
// row-major, Y f32.  Each wave computes NT consecutive 16x16 n-tiles sharing
// one A fragment.  8 waves / 256-thread block.
// M,N multiples of 16; K multiple of 32; (N/16) % NT == 0.
// ---------------------------------------------------------------------------
template <int NT>
__global__ __launch_bounds__(256)
void gemm_wmma_kernel(const __bf16* __restrict__ X, const __bf16* __restrict__ W,
                      float* __restrict__ Y, int M, int K, int N) {
    const int lane = threadIdx.x & 31;
    const int wave = threadIdx.x >> 5;
    const int ntN  = N >> 4;
    const int grpN = ntN / NT;                 // n-tile groups per row
    const int ngroups = (M >> 4) * grpN;
    const int grp = blockIdx.x * 8 + wave;
    if (grp >= ngroups) return;
    const int tm  = (grp / grpN) << 4;
    const int tn0 = (grp % grpN) * (16 * NT);
    const int half = lane >> 4;      // 0 or 1
    const int lidx = lane & 15;
    const int m = tm + lidx;         // A row for this lane

    v8f acc[NT];
#pragma unroll
    for (int u = 0; u < NT; ++u) acc[u] = {};

    for (int kc = 0; kc < K; kc += 32) {
        // ---- A fragment: contiguous k order (same as B) ----
        // lane (half,lidx): element e holds k = half*16 + e
        const __bf16* ap = X + (size_t)m * K + kc + half * 16;
        bf16x8 r0 = *(const bf16x8*)(ap);
        bf16x8 r1 = *(const bf16x8*)(ap + 8);
        v16bf a = __builtin_shufflevector(r0, r1,
                                          0, 1, 2, 3, 4, 5, 6, 7,
                                          8, 9, 10, 11, 12, 13, 14, 15);

        // ---- B fragments: contiguous k order (same as A) ----
        // lane (half,lidx): element e holds k = half*16 + e, col n
        v16bf b[NT];
#pragma unroll
        for (int u = 0; u < NT; ++u) {
            const int n = tn0 + u * 16 + lidx;
            b[u] = *(const v16bf*)(W + (size_t)n * K + kc + half * 16);
        }

#pragma unroll
        for (int u = 0; u < NT; ++u)
            acc[u] = __builtin_amdgcn_wmma_f32_16x16x32_bf16(
                false, a, false, b[u], (short)0, acc[u], false, false);
    }

    // ---- store: 32-bit C/D 16x16 layout: VGPR j -> row tm + half*8 + j, col n
#pragma unroll
    for (int u = 0; u < NT; ++u) {
        const int n = tn0 + u * 16 + lidx;
#pragma unroll
        for (int j = 0; j < 8; ++j)
            Y[(size_t)(tm + half * 8 + j) * N + n] = acc[u][j];
    }
}

// ---------------------------------------------------------------------------
// f32 -> bf16 conversion (plain) and convert+transpose (K,N)f32 -> (N,K)bf16
// ---------------------------------------------------------------------------
__global__ __launch_bounds__(256)
void cvt_kernel(const float* __restrict__ src, __bf16* __restrict__ dst, int total) {
    int i = blockIdx.x * blockDim.x + threadIdx.x;
    if (i < total) dst[i] = f2bf(src[i]);
}

__global__ __launch_bounds__(256)
void cvtT_kernel(const float* __restrict__ src, __bf16* __restrict__ dst,
                 int K, int N) {
    int i = blockIdx.x * blockDim.x + threadIdx.x;
    if (i >= K * N) return;
    int k = i / N, n = i % N;
    dst[(size_t)n * K + k] = f2bf(src[i]);
}

// ---------------------------------------------------------------------------
// Token shift + six channel mixes (outputs bf16: GEMM operands only)
// ---------------------------------------------------------------------------
__global__ __launch_bounds__(256)
void mix_kernel(const float* __restrict__ x,
                const float* __restrict__ cr, const float* __restrict__ cw,
                const float* __restrict__ ck, const float* __restrict__ cv,
                const float* __restrict__ ca, const float* __restrict__ cg,
                __bf16* __restrict__ xr, __bf16* __restrict__ xw,
                __bf16* __restrict__ xk, __bf16* __restrict__ xv,
                __bf16* __restrict__ xa, __bf16* __restrict__ xg) {
    size_t idx = (size_t)blockIdx.x * blockDim.x + threadIdx.x;
    if (idx >= (size_t)BT * CC) return;
    int c = (int)(idx % CC);
    int t = (int)((idx / CC) % TT);
    float xv_ = x[idx];
    float prev = (t == 0) ? 0.0f : x[idx - CC];
    float xx = prev - xv_;
    xr[idx] = f2bf(xv_ + xx * cr[c]);
    xw[idx] = f2bf(xv_ + xx * cw[c]);
    xk[idx] = f2bf(xv_ + xx * ck[c]);
    xv[idx] = f2bf(xv_ + xx * cv[c]);
    xa[idx] = f2bf(xv_ + xx * ca[c]);
    xg[idx] = f2bf(xv_ + xx * cg[c]);
}

__global__ __launch_bounds__(256)
void tanh_bf_kernel(const float* __restrict__ h, __bf16* __restrict__ o, int total) {
    int i = blockIdx.x * blockDim.x + threadIdx.x;
    if (i < total) o[i] = f2bf(tanhf(h[i]));
}

__global__ __launch_bounds__(256)
void sigmoid_bf_kernel(const float* __restrict__ h, __bf16* __restrict__ o, int total) {
    int i = blockIdx.x * blockDim.x + threadIdx.x;
    if (i < total) o[i] = f2bf(sigmoidf_(h[i]));
}

// dec = exp(-exp(-softplus(-(w0 + y)) - 0.5))
__global__ __launch_bounds__(256)
void ew_dec_kernel(float* __restrict__ y, const float* __restrict__ w0) {
    size_t idx = (size_t)blockIdx.x * blockDim.x + threadIdx.x;
    if (idx >= (size_t)BT * CC) return;
    int c = (int)(idx % CC);
    float w = -log1pf(expf(-(w0[c] + y[idx]))) - 0.5f;
    y[idx] = expf(-expf(w));
}

// a = sigmoid(a0 + y)
__global__ __launch_bounds__(256)
void ew_asig_kernel(float* __restrict__ y, const float* __restrict__ a0) {
    size_t idx = (size_t)blockIdx.x * blockDim.x + threadIdx.x;
    if (idx >= (size_t)BT * CC) return;
    int c = (int)(idx % CC);
    y[idx] = sigmoidf_(a0[c] + y[idx]);
}

// v = v_raw + (v_first - v_raw) * sigmoid(v0 + y)
__global__ __launch_bounds__(256)
void ew_vmix_kernel(float* __restrict__ v, const float* __restrict__ v_first,
                    const float* __restrict__ y, const float* __restrict__ v0) {
    size_t idx = (size_t)blockIdx.x * blockDim.x + threadIdx.x;
    if (idx >= (size_t)BT * CC) return;
    int c = (int)(idx % CC);
    float s = sigmoidf_(v0[c] + y[idx]);
    float vr = v[idx];
    v[idx] = vr + (v_first[idx] - vr) * s;
}

// ---------------------------------------------------------------------------
// kk normalization + (-kk), b = kk*a, k modification.  One wave per (token,head)
// ---------------------------------------------------------------------------
__global__ __launch_bounds__(256)
void kk_kernel(float* __restrict__ k,          // in: k_raw, out: k_mod (in place)
               const float* __restrict__ a,    // sigmoid(a)
               const float* __restrict__ k_k, const float* __restrict__ k_a,
               float* __restrict__ negkk, float* __restrict__ bbuf) {
    int lane = threadIdx.x & 31, wave = threadIdx.x >> 5;
    int group = blockIdx.x * 8 + wave;
    if (group >= BT * HH) return;
    int h = group % HH;
    size_t base = (size_t)(group / HH) * CC + h * NN;
    int n0 = lane, n1 = lane + 32;
    int c0 = h * NN + n0, c1 = h * NN + n1;
    float kr0 = k[base + n0], kr1 = k[base + n1];
    float kk0 = kr0 * k_k[c0], kk1 = kr1 * k_k[c1];
    float ss = wave_sum(kk0 * kk0 + kk1 * kk1);
    float inv = 1.0f / fmaxf(sqrtf(ss), 1e-12f);
    kk0 *= inv; kk1 *= inv;
    float a0 = a[base + n0], a1 = a[base + n1];
    negkk[base + n0] = -kk0;           negkk[base + n1] = -kk1;
    bbuf[base + n0]  = kk0 * a0;       bbuf[base + n1]  = kk1 * a1;
    k[base + n0] = kr0 * (1.0f + (a0 - 1.0f) * k_a[c0]);
    k[base + n1] = kr1 * (1.0f + (a1 - 1.0f) * k_a[c1]);
}

// ---------------------------------------------------------------------------
// WKV7 recurrent scan, both directions. One workgroup (64 threads) per
// (batch, head, dir); each thread owns one 64-wide state row in registers.
// dir=1 iterates t = T-1..0 and stores y at t, which equals flip(x_b).
// global_prefetch of step t+1 operands hides load latency behind the FMA body.
// ---------------------------------------------------------------------------
__global__ __launch_bounds__(64)
void wkv_scan_kernel(const float* __restrict__ r, const float* __restrict__ dec,
                     const float* __restrict__ k, const float* __restrict__ v,
                     const float* __restrict__ na, const float* __restrict__ bb,
                     float* __restrict__ ysf, float* __restrict__ ysb) {
    const int inst = blockIdx.x;        // 0..2*B*H-1
    const int dir = inst >> 5;          // B*H = 32 per direction
    const int bh = inst & 31;
    const int b = bh >> 3, h = bh & 7;
    const int tid = threadIdx.x;        // state row i

    float S[NN];
#pragma unroll
    for (int j = 0; j < NN; ++j) S[j] = 0.0f;

    __shared__ float sr[NN], sd[NN], sk[NN], sv[NN], sa[NN], sb[NN];
    float* yout = dir ? ysb : ysf;

    for (int tt = 0; tt < TT; ++tt) {
        int t = dir ? (TT - 1 - tt) : tt;
        size_t base = ((size_t)b * TT + t) * CC + h * NN;
        sr[tid] = r[base + tid];  sd[tid] = dec[base + tid];
        sk[tid] = k[base + tid];  sv[tid] = v[base + tid];
        sa[tid] = na[base + tid]; sb[tid] = bb[base + tid];

        // prefetch next step's operands (global_prefetch_b8)
        if (tt + 1 < TT) {
            int t2 = dir ? (TT - 2 - tt) : (tt + 1);
            size_t b2 = ((size_t)b * TT + t2) * CC + h * NN + tid;
            __builtin_prefetch(&r[b2], 0, 0);
            __builtin_prefetch(&dec[b2], 0, 0);
            __builtin_prefetch(&k[b2], 0, 0);
            __builtin_prefetch(&v[b2], 0, 0);
            __builtin_prefetch(&na[b2], 0, 0);
            __builtin_prefetch(&bb[b2], 0, 0);
        }
        __syncthreads();

        float sai = 0.0f;
#pragma unroll
        for (int j = 0; j < NN; ++j) sai += S[j] * sa[j];
        float vi = sv[tid];
        float y = 0.0f;
#pragma unroll
        for (int j = 0; j < NN; ++j) {
            float s = S[j] * sd[j] + sai * sb[j] + vi * sk[j];
            S[j] = s;
            y += s * sr[j];
        }
        yout[base + tid] = y;
        __syncthreads();
    }
}

// ---------------------------------------------------------------------------
// Gate combine + per-head GroupNorm + residual + g multiply.
// One wave per (token, head); lane handles n and n+32. Output bf16 (feeds Wo).
// ---------------------------------------------------------------------------
__global__ __launch_bounds__(256)
void combine_kernel(const float* __restrict__ x,
                    const float* __restrict__ ysf, const float* __restrict__ ysb,
                    const float* __restrict__ gate_w,
                    const float* __restrict__ ln_g, const float* __restrict__ ln_b,
                    const float* __restrict__ r, const float* __restrict__ kmod,
                    const float* __restrict__ r_k, const float* __restrict__ v,
                    const float* __restrict__ g, __bf16* __restrict__ out) {
    int lane = threadIdx.x & 31, wave = threadIdx.x >> 5;
    int group = blockIdx.x * 8 + wave;
    if (group >= BT * HH) return;
    int h = group % HH;
    int bt = group / HH;
    int t = bt % TT;
    size_t base = (size_t)bt * CC + h * NN;
    int n0 = lane, n1 = lane + 32;
    int c0 = h * NN + n0, c1 = h * NN + n1;

    // token-shift diff (recomputed) -> gate logit
    float x0 = x[base + n0], x1 = x[base + n1];
    float p0 = (t == 0) ? 0.0f : x[base + n0 - CC];
    float p1 = (t == 0) ? 0.0f : x[base + n1 - CC];
    float xx0 = p0 - x0, xx1 = p1 - x1;
    float gd = wave_sum(xx0 * gate_w[c0] + xx1 * gate_w[c1]);
    float gate = sigmoidf_(gd);

    float xo0 = gate * ysf[base + n0] + (1.0f - gate) * ysb[base + n0];
    float xo1 = gate * ysf[base + n1] + (1.0f - gate) * ysb[base + n1];

    float s  = wave_sum(xo0 + xo1);
    float s2 = wave_sum(xo0 * xo0 + xo1 * xo1);
    float mean = s * (1.0f / NN);
    float var = s2 * (1.0f / NN) - mean * mean;
    float rstd = rsqrtf(var + EPS_GN);
    float xn0 = (xo0 - mean) * rstd * ln_g[c0] + ln_b[c0];
    float xn1 = (xo1 - mean) * rstd * ln_g[c1] + ln_b[c1];

    float rk = wave_sum(r[base + n0] * kmod[base + n0] * r_k[c0] +
                        r[base + n1] * kmod[base + n1] * r_k[c1]);
    xn0 += rk * v[base + n0];
    xn1 += rk * v[base + n1];

    out[base + n0] = f2bf(xn0 * g[base + n0]);
    out[base + n1] = f2bf(xn1 * g[base + n1]);
}

// ---------------------------------------------------------------------------
// Launcher
// ---------------------------------------------------------------------------
extern "C" void kernel_launch(void* const* d_in, const int* in_sizes, int n_in,
                              void* d_out, int out_size, void* d_ws, size_t ws_size,
                              hipStream_t stream) {
    (void)in_sizes; (void)n_in; (void)out_size; (void)ws_size;
    const float* x       = (const float*)d_in[0];
    const float* v_first = (const float*)d_in[1];
    const float* x_r = (const float*)d_in[2];
    const float* x_w = (const float*)d_in[3];
    const float* x_k = (const float*)d_in[4];
    const float* x_v = (const float*)d_in[5];
    const float* x_a = (const float*)d_in[6];
    const float* x_g = (const float*)d_in[7];
    const float* w0  = (const float*)d_in[8];
    const float* w1  = (const float*)d_in[9];
    const float* w2  = (const float*)d_in[10];
    const float* a0  = (const float*)d_in[11];
    const float* a1  = (const float*)d_in[12];
    const float* a2  = (const float*)d_in[13];
    const float* v0  = (const float*)d_in[14];
    const float* v1  = (const float*)d_in[15];
    const float* v2  = (const float*)d_in[16];
    const float* g1  = (const float*)d_in[17];
    const float* g2  = (const float*)d_in[18];
    const float* k_k = (const float*)d_in[19];
    const float* k_a = (const float*)d_in[20];
    const float* r_k = (const float*)d_in[21];
    const float* gate_w = (const float*)d_in[22];
    const float* ln_g = (const float*)d_in[23];
    const float* ln_b = (const float*)d_in[24];
    const float* Wr = (const float*)d_in[25];
    const float* Wk = (const float*)d_in[26];
    const float* Wv = (const float*)d_in[27];
    const float* Wo = (const float*)d_in[28];
    float* out = (float*)d_out;

    const size_t BTC = (size_t)BT * CC;   // 2,097,152

    // ---- workspace carve (256B aligned slots) ----
    char* p = (char*)d_ws;
    auto carve = [&](size_t bytes) -> void* {
        void* q = (void*)p;
        p += (bytes + 255) & ~(size_t)255;
        return q;
    };
    // f32 buffers
    float* rb    = (float*)carve(BTC * 4);
    float* decb  = (float*)carve(BTC * 4);
    float* kb    = (float*)carve(BTC * 4);
    float* vb    = (float*)carve(BTC * 4);
    float* ab    = (float*)carve(BTC * 4);
    float* gb    = (float*)carve(BTC * 4);
    float* negkk = (float*)carve(BTC * 4);
    float* bbuf  = (float*)carve(BTC * 4);
    float* ysf   = (float*)carve(BTC * 4);
    float* ysb   = (float*)carve(BTC * 4);
    float* hidf  = (float*)carve((size_t)BT * 128 * 4);
    // bf16 activation buffers
    __bf16* xrb  = (__bf16*)carve(BTC * 2);
    __bf16* xwb  = (__bf16*)carve(BTC * 2);
    __bf16* xkb  = (__bf16*)carve(BTC * 2);
    __bf16* xvb  = (__bf16*)carve(BTC * 2);
    __bf16* xab  = (__bf16*)carve(BTC * 2);
    __bf16* xgb  = (__bf16*)carve(BTC * 2);
    __bf16* xbf  = (__bf16*)carve(BTC * 2);   // bf16 copy of x (v path)
    __bf16* xogb = (__bf16*)carve(BTC * 2);   // combine output (feeds Wo)
    __bf16* hidb = (__bf16*)carve((size_t)BT * 128 * 2);
    // bf16 weights, all laid out (N,K) row-major
    __bf16* WrB = (__bf16*)carve((size_t)CC * CC * 2);
    __bf16* WkB = (__bf16*)carve((size_t)CC * CC * 2);
    __bf16* WvB = (__bf16*)carve((size_t)CC * CC * 2);
    __bf16* WoB = (__bf16*)carve((size_t)CC * CC * 2);
    __bf16* w1T = (__bf16*)carve((size_t)64 * CC * 2);
    __bf16* w2T = (__bf16*)carve((size_t)CC * 64 * 2);
    __bf16* a1T = (__bf16*)carve((size_t)64 * CC * 2);
    __bf16* a2T = (__bf16*)carve((size_t)CC * 64 * 2);
    __bf16* v1T = (__bf16*)carve((size_t)32 * CC * 2);
    __bf16* v2T = (__bf16*)carve((size_t)CC * 32 * 2);
    __bf16* g1T = (__bf16*)carve((size_t)128 * CC * 2);
    __bf16* g2T = (__bf16*)carve((size_t)CC * 128 * 2);

    const int EW_BLOCKS = (int)((BTC + 255) / 256);
    auto cvt = [&](const float* s, __bf16* d, int total) {
        cvt_kernel<<<(total + 255) / 256, 256, 0, stream>>>(s, d, total);
    };
    auto cvtT = [&](const float* s, __bf16* d, int K, int N) {
        cvtT_kernel<<<(K * N + 255) / 256, 256, 0, stream>>>(s, d, K, N);
    };
    auto gemm = [&](const __bf16* Xp, const __bf16* Wp, float* Yp, int M, int K, int N) {
        if (N % 64 == 0) {
            int groups = (M >> 4) * (N >> 6);
            gemm_wmma_kernel<4><<<(groups + 7) / 8, 256, 0, stream>>>(Xp, Wp, Yp, M, K, N);
        } else { // N == 32
            int groups = (M >> 4) * (N >> 5);
            gemm_wmma_kernel<2><<<(groups + 7) / 8, 256, 0, stream>>>(Xp, Wp, Yp, M, K, N);
        }
    };

    // 0) weight conversions (big projections already (N,K); MLP weights transposed)
    cvt(Wr, WrB, CC * CC);
    cvt(Wk, WkB, CC * CC);
    cvt(Wv, WvB, CC * CC);
    cvt(Wo, WoB, CC * CC);
    cvtT(w1, w1T, CC, 64);
    cvtT(w2, w2T, 64, CC);
    cvtT(a1, a1T, CC, 64);
    cvtT(a2, a2T, 64, CC);
    cvtT(v1, v1T, CC, 32);
    cvtT(v2, v2T, 32, CC);
    cvtT(g1, g1T, CC, 128);
    cvtT(g2, g2T, 128, CC);
    cvt(x, xbf, (int)BTC);

    // 1) token shift + mixes (bf16 outputs)
    mix_kernel<<<EW_BLOCKS, 256, 0, stream>>>(x, x_r, x_w, x_k, x_v, x_a, x_g,
                                              xrb, xwb, xkb, xvb, xab, xgb);

    // 2) big projections (Y = X @ W^T)
    gemm(xrb, WrB, rb, BT, CC, CC);
    gemm(xkb, WkB, kb, BT, CC, CC);
    gemm(xvb, WvB, vb, BT, CC, CC);

    // 3) w path: dec = exp(-exp(-softplus(-(w0 + tanh(xw@w1)@w2)) - 0.5))
    gemm(xwb, w1T, hidf, BT, CC, 64);
    tanh_bf_kernel<<<(BT * 64 + 255) / 256, 256, 0, stream>>>(hidf, hidb, BT * 64);
    gemm(hidb, w2T, decb, BT, 64, CC);
    ew_dec_kernel<<<EW_BLOCKS, 256, 0, stream>>>(decb, w0);

    // 4) a path: a = sigmoid(a0 + (xa@a1)@a2)
    gemm(xab, a1T, hidf, BT, CC, 64);
    cvt(hidf, hidb, BT * 64);
    gemm(hidb, a2T, ab, BT, 64, CC);
    ew_asig_kernel<<<EW_BLOCKS, 256, 0, stream>>>(ab, a0);

    // 5) v path: v = v_raw + (v_first - v_raw) * sigmoid(v0 + (x@v1)@v2)
    gemm(xbf, v1T, hidf, BT, CC, 32);
    cvt(hidf, hidb, BT * 32);
    gemm(hidb, v2T, bbuf, BT, 32, CC);  // bbuf as temp (overwritten by kk later)
    ew_vmix_kernel<<<EW_BLOCKS, 256, 0, stream>>>(vb, v_first, bbuf, v0);

    // 6) g path: g = sigmoid(xg@g1) @ g2
    gemm(xgb, g1T, hidf, BT, CC, 128);
    sigmoid_bf_kernel<<<(BT * 128 + 255) / 256, 256, 0, stream>>>(hidf, hidb, BT * 128);
    gemm(hidb, g2T, gb, BT, 128, CC);

    // 7) kk normalize, -kk, b = kk*a, k modification
    kk_kernel<<<(BT * HH + 7) / 8, 256, 0, stream>>>(kb, ab, k_k, k_a, negkk, bbuf);

    // 8) bidirectional WKV7 scan (64 instances, 64 threads each)
    wkv_scan_kernel<<<2 * BB * HH, 64, 0, stream>>>(rb, decb, kb, vb, negkk, bbuf,
                                                    ysf, ysb);

    // 9) gate + groupnorm + residual + g  (bf16 output feeds Wo)
    combine_kernel<<<(BT * HH + 7) / 8, 256, 0, stream>>>(
        x, ysf, ysb, gate_w, ln_g, ln_b, rb, kb, r_k, vb, gb, xogb);

    // 10) output projection: out = (xo*g) @ Wo^T
    gemm(xogb, WoB, out, BT, CC, CC);

    // 11) second tuple output: v_first passthrough
    hipMemcpyAsync(out + BTC, v_first, BTC * sizeof(float),
                   hipMemcpyDeviceToDevice, stream);
}
